// LSTM_submodule_47888885350825
// MI455X (gfx1250) — matrix operand order repaired and checked by reading, not compiled, not run on hardware
//
#include <hip/hip_runtime.h>
#include <hip/hip_bf16.h>

typedef __attribute__((ext_vector_type(16))) _Float16 v16h;
typedef __attribute__((ext_vector_type(8)))  float    v8f;

#define S_TOTAL 19648   // B*N = 64*307 sequential LSTM steps
#define TB      12      // LSTM "batch" (T_IN)
#define FIN     3
#define HIDN    128
#define HSTR    136     // LDS row stride (f16 units), padded vs 128 to stagger banks

// Fast transcendentals: v_exp_f32 + v_rcp_f32, no IEEE div expansion.
__device__ __forceinline__ float rcpf_(float x) { return __builtin_amdgcn_rcpf(x); }
__device__ __forceinline__ float sigmf_(float x) { return rcpf_(1.0f + __expf(-x)); }
__device__ __forceinline__ float tanhf_(float x) { return 1.0f - 2.0f * rcpf_(__expf(2.0f * x) + 1.0f); }

// One persistent workgroup (8 wave32 = one WGP) runs the whole 19648-step scan.
// Wave w owns hidden columns [16w, 16w+16) for ALL 4 gates -> the four gate
// accumulators for a hidden channel live in the same lane; cell state c stays
// in registers for the entire scan. W_hh is held as 16 f16 WMMA B-fragments
// per wave in VGPRs (loaded/converted once). h crosses waves through a
// double-buffered LDS tile.
__global__ __launch_bounds__(256, 1)
void lstm_scan_kernel(const float* __restrict__ xg,
                      const float* __restrict__ W_ih,
                      const float* __restrict__ W_hh,
                      const float* __restrict__ b_ih,
                      const float* __restrict__ b_hh,
                      const float* __restrict__ W_lin,
                      const float* __restrict__ b_lin,
                      float* __restrict__ outg)
{
    __shared__ _Float16 hbuf[2][16 * HSTR];   // h, rows 0..15 (12..15 stay zero)
    __shared__ float    xbuf[2][TB * FIN];    // x for current / next step
    __shared__ float    opart[8][TB];         // per-wave output partials

    const int tid  = threadIdx.x;
    const int w    = tid >> 5;     // wave id 0..7
    const int lane = tid & 31;
    const int nl   = lane & 15;    // column-within-tile / row id
    const int hl   = lane >> 4;    // half-wave select
    const int jcol = w * 16 + nl;  // this lane's hidden column (0..127)

    // ---- init: zero h buffers (rows 12..15 must be zero forever) ----
    for (int i = tid; i < 2 * 16 * HSTR; i += 256)
        ((_Float16*)hbuf)[i] = (_Float16)0.0f;
    if (tid < TB * FIN) {                       // x[0]: layout (s, f, t) -> xbuf[t*3+f]
        int t = tid / 3, f = tid % 3;
        xbuf[0][tid] = xg[f * TB + t];
    }

    // ---- per-lane constant preload (once) ----
    float bsum[4], wih[4][3];
    v16h  bfrag[4][4];   // [gate][k-block] : B = W_hh^T tile, column n = jcol+gate*128
    #pragma unroll
    for (int j = 0; j < 4; ++j) {
        const int n = j * HIDN + jcol;
        bsum[j]   = b_ih[n] + b_hh[n];
        wih[j][0] = W_ih[n * 3 + 0];
        wih[j][1] = W_ih[n * 3 + 1];
        wih[j][2] = W_ih[n * 3 + 2];
        const float* wr = W_hh + (size_t)n * HIDN;   // B[k][n] = W_hh[n][k]
        #pragma unroll
        for (int kb = 0; kb < 4; ++kb) {
            const int kbase = kb * 32 + hl * 16;     // contiguous k order: half-wave = K half
            #pragma unroll
            for (int v = 0; v < 8; ++v) {
                bfrag[j][kb][2 * v]     = (_Float16)wr[kbase + 2 * v];
                bfrag[j][kb][2 * v + 1] = (_Float16)wr[kbase + 2 * v + 1];
            }
        }
    }
    const float wlin = W_lin[jcol];
    const float blin = b_lin[0];

    v8f cst = {};   // cell state, rows (r + 8*hl) of this lane's hidden column
    __syncthreads();

    for (int s = 0; s < S_TOTAL; ++s) {
        const int cur = s & 1, nxt = cur ^ 1;

        // x rows needed by this lane's accumulator rows (m = r + 8*hl)
        float xr[8][3];
        #pragma unroll
        for (int r = 0; r < 8; ++r) {
            int m = r + 8 * hl; if (m >= TB) m = 0;     // padded rows: any finite value
            xr[r][0] = xbuf[cur][m * 3 + 0];
            xr[r][1] = xbuf[cur][m * 3 + 1];
            xr[r][2] = xbuf[cur][m * 3 + 2];
        }

        // accumulators initialized with gx = x @ W_ih^T + b_ih + b_hh (F_IN=3 FMAs)
        v8f acc[4];
        #pragma unroll
        for (int j = 0; j < 4; ++j)
            #pragma unroll
            for (int r = 0; r < 8; ++r)
                acc[j][r] = bsum[j] + xr[r][0] * wih[j][0]
                                    + xr[r][1] * wih[j][1]
                                    + xr[r][2] * wih[j][2];

        // z += h @ W_hh^T : 4 k-blocks x 4 gate tiles of v_wmma_f32_16x16x32_f16
        const _Float16* hb = hbuf[cur];
        #pragma unroll
        for (int kb = 0; kb < 4; ++kb) {
            v16h a;   // A 16x32, contiguous k order: half-wave = K half (matches B)
            #pragma unroll
            for (int v = 0; v < 8; ++v) {
                const int k0 = kb * 32 + hl * 16 + 2 * v;
                a[2 * v]     = hb[nl * HSTR + k0];
                a[2 * v + 1] = hb[nl * HSTR + k0 + 1];
            }
            #pragma unroll
            for (int j = 0; j < 4; ++j)
                acc[j] = __builtin_amdgcn_wmma_f32_16x16x32_f16(
                    false, a, false, bfrag[j][kb], (short)0, acc[j], false, false);
        }

        // gates + state update, all in registers (v_exp_f32 / v_rcp_f32 only)
        float pr[8];
        _Float16* hn = hbuf[nxt];
        #pragma unroll
        for (int r = 0; r < 8; ++r) {
            const float iv = sigmf_(acc[0][r]);
            const float fv = sigmf_(acc[1][r]);
            const float gv = tanhf_(acc[2][r]);
            const float ov = sigmf_(acc[3][r]);
            const float cn = fv * cst[r] + iv * gv;
            cst[r] = cn;
            const float hv = ov * tanhf_(cn);
            const int m = r + 8 * hl;
            if (m < TB) hn[m * HSTR + jcol] = (_Float16)hv;
            pr[r] = hv * wlin;
        }

        // reduce output partials over this wave's 16 hidden columns (within half)
        #pragma unroll
        for (int r = 0; r < 8; ++r) {
            float p = pr[r];
            p += __shfl_xor(p, 1, 32);
            p += __shfl_xor(p, 2, 32);
            p += __shfl_xor(p, 4, 32);
            p += __shfl_xor(p, 8, 32);
            pr[r] = p;
        }
        if (nl == 0) {
            #pragma unroll
            for (int r = 0; r < 8; ++r) {
                const int m = r + 8 * hl;
                if (m < TB) opart[w][m] = pr[r];
            }
        }
        __syncthreads();

        // epilogue: finalize out[s], prefetch x[s+1] into the other buffer
        if (tid < TB) {
            float o = blin;
            #pragma unroll
            for (int ww = 0; ww < 8; ++ww) o += opart[ww][tid];
            outg[(size_t)s * TB + tid] = o;
        }
        const int q = tid - 32;                         // threads 32..67 cover 36 loads
        if (q >= 0 && q < TB * FIN && (s + 1) < S_TOTAL) {
            const int t = q / 3, f = q % 3;
            xbuf[nxt][q] = xg[(size_t)(s + 1) * (TB * FIN) + f * TB + t];
        }
        __syncthreads();
    }
}

extern "C" void kernel_launch(void* const* d_in, const int* in_sizes, int n_in,
                              void* d_out, int out_size, void* d_ws, size_t ws_size,
                              hipStream_t stream) {
    (void)in_sizes; (void)n_in; (void)out_size; (void)d_ws; (void)ws_size;
    const float* x     = (const float*)d_in[0];
    const float* W_ih  = (const float*)d_in[1];
    const float* W_hh  = (const float*)d_in[2];
    const float* b_ih  = (const float*)d_in[3];
    const float* b_hh  = (const float*)d_in[4];
    const float* W_lin = (const float*)d_in[5];
    const float* b_lin = (const float*)d_in[6];
    float* out = (float*)d_out;
    // Strictly sequential recurrence over 19648 steps -> one persistent workgroup.
    lstm_scan_kernel<<<1, 256, 0, stream>>>(x, W_ih, W_hh, b_ih, b_hh, W_lin, b_lin, out);
}
